// MixtureOfExperts_72438918414751
// MI455X (gfx1250) — compile-verified, matrix operand order checked by reading, not run on hardware
//
#include <hip/hip_runtime.h>
#include <hip/hip_bf16.h>

// Problem dims (fixed by the reference)
constexpr int Btok = 4096;   // B*S tokens
constexpr int Hdim = 1024;
constexpr int Fdim = 4096;
constexpr int NE   = 8;

typedef __attribute__((ext_vector_type(16))) __bf16 v16bf;
typedef __attribute__((ext_vector_type(8)))  float  v8f;

// ---- f32 -> bf16 (round to nearest even) ----
__device__ __forceinline__ unsigned short f2bf(float f) {
    unsigned u = __builtin_bit_cast(unsigned, f);
    u += 0x7FFFu + ((u >> 16) & 1u);
    return (unsigned short)(u >> 16);
}
__device__ __forceinline__ unsigned pack2(float a, float b) {
    return (unsigned)f2bf(a) | ((unsigned)f2bf(b) << 16);
}
// fast silu: g * v_rcp_f32(1 + e^-g)  (no IEEE div chain)
__device__ __forceinline__ float fast_silu(float g) {
    return g * __builtin_amdgcn_rcpf(1.f + __expf(-g));
}

// Load a 16x32 bf16 A/B fragment from LDS (row-major, `stride` elems per row)
// per the CDNA5 16-bit A 16x32 VGPR layout:
//   lanes 0-15  : row = lane,    K =  0..7 (v0-3), 16..23 (v4-7)
//   lanes 16-31 : row = lane-16, K =  8..15(v0-3), 24..31 (v4-7)
__device__ __forceinline__ v16bf load_frag(const unsigned short* base, int stride, int lane) {
    int r  = lane & 15;
    int k0 = (lane >> 4) * 8;
    const unsigned short* p = base + r * stride + k0;
    union { uint4 q[2]; v16bf v; } u;
    u.q[0] = *(const uint4*)(p);
    u.q[1] = *(const uint4*)(p + 16);
    return u.v;
}

__device__ __forceinline__ v8f wmma_bf16(v16bf a, v16bf b, v8f c) {
    return __builtin_amdgcn_wmma_f32_16x16x32_bf16(false, a, false, b, (short)0, c, false, false);
}

// ---------------- Prep: x f32 -> bf16 (elementwise) ----------------
__global__ void __launch_bounds__(256)
convert_x(const float* __restrict__ src, unsigned short* __restrict__ dst) {
    int i = blockIdx.x * blockDim.x + threadIdx.x;   // 8 floats per thread
    const float* s = src + (size_t)i * 8;
    float4 a = *(const float4*)(s);
    float4 b = *(const float4*)(s + 4);
    uint4 q = { pack2(a.x, a.y), pack2(a.z, a.w), pack2(b.x, b.y), pack2(b.z, b.w) };
    *(uint4*)&dst[(size_t)i * 8] = q;
}

// ---------------- Prep: transpose+convert [mat][R][C] f32 -> [mat][C][R] bf16 ----------------
__global__ void __launch_bounds__(256)
transpose_cvt(const float* __restrict__ src, unsigned short* __restrict__ dst,
              int R, int C) {
    __shared__ unsigned short tile[64][72];          // 72: keep 16B row alignment, de-bank
    const int mat = blockIdx.z;
    const int r0 = blockIdx.y * 64, c0 = blockIdx.x * 64;
    {
        int rr = threadIdx.x >> 2, cq = (threadIdx.x & 3) * 16;
        const float* sp = src + ((size_t)mat * R + r0 + rr) * C + c0 + cq;
        float4 v0 = *(const float4*)(sp);
        float4 v1 = *(const float4*)(sp + 4);
        float4 v2 = *(const float4*)(sp + 8);
        float4 v3 = *(const float4*)(sp + 12);
        uint4 q0 = { pack2(v0.x, v0.y), pack2(v0.z, v0.w), pack2(v1.x, v1.y), pack2(v1.z, v1.w) };
        uint4 q1 = { pack2(v2.x, v2.y), pack2(v2.z, v2.w), pack2(v3.x, v3.y), pack2(v3.z, v3.w) };
        *(uint4*)&tile[rr][cq]     = q0;
        *(uint4*)&tile[rr][cq + 8] = q1;
    }
    __syncthreads();
    {
        int wc = threadIdx.x >> 2, rq = (threadIdx.x & 3) * 16;
        unsigned o[8];
#pragma unroll
        for (int j = 0; j < 8; ++j)
            o[j] = (unsigned)tile[rq + 2 * j][wc] | ((unsigned)tile[rq + 2 * j + 1][wc] << 16);
        unsigned short* d = dst + ((size_t)mat * C + c0 + wc) * R + r0 + rq;
        *(uint4*)&d[0] = *(uint4*)&o[0];
        *(uint4*)&d[8] = *(uint4*)&o[4];
    }
}

// ---------------- Gating: logits -> softmax -> top-2 -> renormalized cw ----------------
__global__ void __launch_bounds__(256)
moe_gate(const float* __restrict__ x, const float* __restrict__ Wg,
         const float* __restrict__ bg, float* __restrict__ cw) {
    __shared__ float Wgs[Hdim * NE];            // 32 KB router weights
    const int tid = threadIdx.x;
    for (int i = tid; i < Hdim * NE / 4; i += 256)
        ((float4*)Wgs)[i] = ((const float4*)Wg)[i];
    __syncthreads();

    int t = blockIdx.x * blockDim.x + tid;
    const float* xr = x + (size_t)t * Hdim;
    float acc[NE];
#pragma unroll
    for (int e = 0; e < NE; ++e) acc[e] = 0.f;
    for (int h = 0; h < Hdim; ++h) {
        float xv = xr[h];
        float4 w0 = *(const float4*)(&Wgs[h * NE]);
        float4 w1 = *(const float4*)(&Wgs[h * NE + 4]);
        acc[0] += xv * w0.x; acc[1] += xv * w0.y; acc[2] += xv * w0.z; acc[3] += xv * w0.w;
        acc[4] += xv * w1.x; acc[5] += xv * w1.y; acc[6] += xv * w1.z; acc[7] += xv * w1.w;
    }
#pragma unroll
    for (int e = 0; e < NE; ++e) acc[e] += bg[e];
    float m = acc[0];
#pragma unroll
    for (int e = 1; e < NE; ++e) m = fmaxf(m, acc[e]);
    float p[NE];
#pragma unroll
    for (int e = 0; e < NE; ++e) p[e] = __expf(acc[e] - m);
    int i1 = 0;
#pragma unroll
    for (int e = 1; e < NE; ++e) if (p[e] > p[i1]) i1 = e;
    int i2 = (i1 == 0) ? 1 : 0;
#pragma unroll
    for (int e = 0; e < NE; ++e) if (e != i1 && p[e] > p[i2]) i2 = e;
    float inv = __builtin_amdgcn_rcpf(p[i1] + p[i2]);
#pragma unroll
    for (int e = 0; e < NE; ++e) {
        float v = (e == i1) ? p[i1] * inv : (e == i2) ? p[i2] * inv : 0.f;
        cw[(size_t)t * NE + e] = v;
    }
}

// ---------------- Fused SwiGLU expert FFN: pre-converted bf16 weights ----------------
// Weights already bf16, K-contiguous: W1t/W3t = [e][f][h], W2t = [e][h][f].
// Staging is pure b128 copy: zero conversion VALU in the hot loop.
__global__ void __launch_bounds__(256)
moe_ffn_bf16(const unsigned short* __restrict__ xbf,
             const unsigned short* __restrict__ W1t,
             const unsigned short* __restrict__ W3t,
             const unsigned short* __restrict__ W2t,
             const float* __restrict__ cw, float* __restrict__ out) {
    constexpr int MT = 64;
    constexpr int FC = 512;
    constexpr int KC = 32;

    __shared__ unsigned short Xs[MT * KC];
    __shared__ unsigned short B1s[64 * KC];
    __shared__ unsigned short B3s[64 * KC];
    __shared__ unsigned short Hs[MT * FC];
    __shared__ unsigned short W2s[128 * KC];
    __shared__ float cws[MT];
    __shared__ int   active;

    const int mtile  = blockIdx.x;
    const int fchunk = blockIdx.y;
    const int e      = blockIdx.z;
    const int tid    = threadIdx.x;
    const int lane   = tid & 31;
    const int wave   = tid >> 5;
    const int mi     = wave & 3;
    const int nh     = wave >> 2;

    if (tid == 0) active = 0;
    __syncthreads();
    if (tid < MT) {
        float v = cw[(size_t)(mtile * MT + tid) * NE + e];
        cws[tid] = v;
        if (v != 0.f) atomicOr(&active, 1);
    }
    __syncthreads();
    if (!active) return;

    const int rbase = (lane >> 4) << 3;
    const int cc    = lane & 15;

    // ---------------- Phase 1 ----------------
    for (int fc = 0; fc < FC; fc += 64) {
        v8f g0 = {}, g1 = {}, u0 = {}, u1 = {};
        for (int k0 = 0; k0 < Hdim; k0 += KC) {
            __syncthreads();
            {   // Xs[64][32] <- xbf (pure b128 copy)
                int r = tid >> 2, c = (tid & 3) * 8;
                *(uint4*)&Xs[r * KC + c] =
                    *(const uint4*)&xbf[(size_t)(mtile * MT + r) * Hdim + k0 + c];
            }
            {   // B1s/B3s [64 f][32 k] <- W1t/W3t (K-contiguous rows, b128 copy)
                int f  = tid >> 2, kq = (tid & 3) * 8;
                size_t rb = ((size_t)e * Fdim + (size_t)fchunk * FC + fc + f) * Hdim + k0 + kq;
                *(uint4*)&B1s[f * KC + kq] = *(const uint4*)&W1t[rb];
                *(uint4*)&B3s[f * KC + kq] = *(const uint4*)&W3t[rb];
                if (k0 + KC < Hdim) {   // global_prefetch_b8 on next K chunk
                    __builtin_prefetch(&W1t[rb + KC], 0, 0);
                    __builtin_prefetch(&W3t[rb + KC], 0, 0);
                }
            }
            __syncthreads();
            v16bf a   = load_frag(&Xs[(mi * 16) * KC], KC, lane);
            v16bf b1a = load_frag(&B1s[((nh * 2 + 0) * 16) * KC], KC, lane);
            v16bf b1b = load_frag(&B1s[((nh * 2 + 1) * 16) * KC], KC, lane);
            v16bf b3a = load_frag(&B3s[((nh * 2 + 0) * 16) * KC], KC, lane);
            v16bf b3b = load_frag(&B3s[((nh * 2 + 1) * 16) * KC], KC, lane);
            g0 = wmma_bf16(a, b1a, g0);
            g1 = wmma_bf16(a, b1b, g1);
            u0 = wmma_bf16(a, b3a, u0);
            u1 = wmma_bf16(a, b3b, u1);
        }
#pragma unroll
        for (int t = 0; t < 2; ++t) {
            v8f g = t ? g1 : g0;
            v8f u = t ? u1 : u0;
            int colb = fc + nh * 32 + t * 16 + cc;
#pragma unroll
            for (int r = 0; r < 8; ++r) {
                int row = mi * 16 + r + rbase;
                float hv = fast_silu(g[r]) * u[r] * cws[row];
                Hs[row * FC + colb] = f2bf(hv);
            }
        }
    }
    __syncthreads();

    // ---------------- Phase 2 ----------------
    for (int nc = 0; nc < Hdim; nc += 128) {
        v8f acc[4] = {};
        for (int k0 = 0; k0 < FC; k0 += KC) {
            __syncthreads();
            {   // W2s [128 n][32 k] <- W2t[e][h][f] (K-contiguous, b128 copies)
                int n = tid >> 1, kq = (tid & 1) * 16;
                size_t rb = ((size_t)e * Hdim + nc + n) * Fdim
                          + (size_t)fchunk * FC + k0 + kq;
                *(uint4*)&W2s[n * KC + kq]     = *(const uint4*)&W2t[rb];
                *(uint4*)&W2s[n * KC + kq + 8] = *(const uint4*)&W2t[rb + 8];
                if (k0 + KC < FC) __builtin_prefetch(&W2t[rb + KC], 0, 0);
            }
            __syncthreads();
            v16bf a = load_frag(&Hs[(mi * 16) * FC + k0], FC, lane);
#pragma unroll
            for (int j = 0; j < 4; ++j) {
                v16bf b = load_frag(&W2s[((nh * 4 + j) * 16) * KC], KC, lane);
                acc[j] = wmma_bf16(a, b, acc[j]);
            }
        }
#pragma unroll
        for (int j = 0; j < 4; ++j)
#pragma unroll
            for (int r = 0; r < 8; ++r) {
                int row = mtile * MT + mi * 16 + r + rbase;
                int col = nc + (nh * 4 + j) * 16 + cc;
                unsafeAtomicAdd(&out[(size_t)row * Hdim + col], acc[j][r]);
            }
    }
}

// ---------------- Fallback FFN: convert f32 weights in-kernel (small ws) ----------------
__global__ void __launch_bounds__(256)
moe_ffn(const float* __restrict__ x,  const float* __restrict__ W1,
        const float* __restrict__ W3, const float* __restrict__ W2,
        const float* __restrict__ cw, float* __restrict__ out) {
    constexpr int MT = 64;
    constexpr int FC = 512;
    constexpr int KC = 32;

    __shared__ unsigned short Xs[MT * KC];
    __shared__ unsigned short B1s[64 * KC];
    __shared__ unsigned short B3s[64 * KC];
    __shared__ unsigned short Hs[MT * FC];
    __shared__ unsigned short W2s[128 * KC];
    __shared__ float cws[MT];
    __shared__ int   active;

    const int mtile  = blockIdx.x;
    const int fchunk = blockIdx.y;
    const int e      = blockIdx.z;
    const int tid    = threadIdx.x;
    const int wave   = tid >> 5;
    const int lane   = tid & 31;
    const int mi     = wave & 3;
    const int nh     = wave >> 2;

    if (tid == 0) active = 0;
    __syncthreads();
    if (tid < MT) {
        float v = cw[(size_t)(mtile * MT + tid) * NE + e];
        cws[tid] = v;
        if (v != 0.f) atomicOr(&active, 1);
    }
    __syncthreads();
    if (!active) return;

    const int rbase = (lane >> 4) << 3;
    const int cc    = lane & 15;
    const int kg    = tid >> 5;
    const int f2    = lane * 2;

    for (int fc = 0; fc < FC; fc += 64) {
        v8f g0 = {}, g1 = {}, u0 = {}, u1 = {};
        for (int k0 = 0; k0 < Hdim; k0 += KC) {
            __syncthreads();
            {
                int r = tid >> 2, c = (tid & 3) * 8;
                const float* src = x + (size_t)(mtile * MT + r) * Hdim + k0 + c;
                float4 fx0 = *(const float4*)(src);
                float4 fx1 = *(const float4*)(src + 4);
                uint4 pk = { pack2(fx0.x, fx0.y), pack2(fx0.z, fx0.w),
                             pack2(fx1.x, fx1.y), pack2(fx1.z, fx1.w) };
                *(uint4*)&Xs[r * KC + c] = pk;
            }
            {
                size_t rb = ((size_t)e * Hdim + (k0 + kg * 4)) * Fdim
                          + (size_t)fchunk * FC + fc + f2;
                float2 a0 = *(const float2*)(W1 + rb);
                float2 a1 = *(const float2*)(W1 + rb + Fdim);
                float2 a2 = *(const float2*)(W1 + rb + 2 * (size_t)Fdim);
                float2 a3 = *(const float2*)(W1 + rb + 3 * (size_t)Fdim);
                float2 b0 = *(const float2*)(W3 + rb);
                float2 b1 = *(const float2*)(W3 + rb + Fdim);
                float2 b2 = *(const float2*)(W3 + rb + 2 * (size_t)Fdim);
                float2 b3 = *(const float2*)(W3 + rb + 3 * (size_t)Fdim);
                uint2 pa0 = { pack2(a0.x, a1.x), pack2(a2.x, a3.x) };
                uint2 pa1 = { pack2(a0.y, a1.y), pack2(a2.y, a3.y) };
                uint2 pb0 = { pack2(b0.x, b1.x), pack2(b2.x, b3.x) };
                uint2 pb1 = { pack2(b0.y, b1.y), pack2(b2.y, b3.y) };
                *(uint2*)&B1s[(f2 + 0) * KC + kg * 4] = pa0;
                *(uint2*)&B1s[(f2 + 1) * KC + kg * 4] = pa1;
                *(uint2*)&B3s[(f2 + 0) * KC + kg * 4] = pb0;
                *(uint2*)&B3s[(f2 + 1) * KC + kg * 4] = pb1;
            }
            __syncthreads();
            v16bf a   = load_frag(&Xs[(mi * 16) * KC], KC, lane);
            v16bf b1a = load_frag(&B1s[((nh * 2 + 0) * 16) * KC], KC, lane);
            v16bf b1b = load_frag(&B1s[((nh * 2 + 1) * 16) * KC], KC, lane);
            v16bf b3a = load_frag(&B3s[((nh * 2 + 0) * 16) * KC], KC, lane);
            v16bf b3b = load_frag(&B3s[((nh * 2 + 1) * 16) * KC], KC, lane);
            g0 = wmma_bf16(a, b1a, g0);
            g1 = wmma_bf16(a, b1b, g1);
            u0 = wmma_bf16(a, b3a, u0);
            u1 = wmma_bf16(a, b3b, u1);
        }
#pragma unroll
        for (int t = 0; t < 2; ++t) {
            v8f g = t ? g1 : g0;
            v8f u = t ? u1 : u0;
            int colb = fc + nh * 32 + t * 16 + cc;
#pragma unroll
            for (int r = 0; r < 8; ++r) {
                int row = mi * 16 + r + rbase;
                float hv = fast_silu(g[r]) * u[r] * cws[row];
                Hs[row * FC + colb] = f2bf(hv);
            }
        }
    }
    __syncthreads();

    const float* W2e = W2 + (size_t)e * Fdim * Hdim + (size_t)fchunk * FC * Hdim;
    for (int nc = 0; nc < Hdim; nc += 128) {
        v8f acc[4] = {};
        for (int k0 = 0; k0 < FC; k0 += KC) {
            __syncthreads();
            {
                const float* src = W2e + (size_t)(k0 + kg * 4) * Hdim + nc;
#pragma unroll
                for (int half = 0; half < 128; half += 64) {
                    const float* s2 = src + half + f2;
                    float2 r0 = *(const float2*)(s2);
                    float2 r1 = *(const float2*)(s2 + Hdim);
                    float2 r2 = *(const float2*)(s2 + 2 * (size_t)Hdim);
                    float2 r3 = *(const float2*)(s2 + 3 * (size_t)Hdim);
                    uint2 p0 = { pack2(r0.x, r1.x), pack2(r2.x, r3.x) };
                    uint2 p1 = { pack2(r0.y, r1.y), pack2(r2.y, r3.y) };
                    *(uint2*)&W2s[(half + f2 + 0) * KC + kg * 4] = p0;
                    *(uint2*)&W2s[(half + f2 + 1) * KC + kg * 4] = p1;
                }
            }
            __syncthreads();
            v16bf a = load_frag(&Hs[(mi * 16) * FC + k0], FC, lane);
#pragma unroll
            for (int j = 0; j < 4; ++j) {
                v16bf b = load_frag(&W2s[((nh * 4 + j) * 16) * KC], KC, lane);
                acc[j] = wmma_bf16(a, b, acc[j]);
            }
        }
#pragma unroll
        for (int j = 0; j < 4; ++j)
#pragma unroll
            for (int r = 0; r < 8; ++r) {
                int row = mtile * MT + mi * 16 + r + rbase;
                int col = nc + (nh * 4 + j) * 16 + cc;
                unsafeAtomicAdd(&out[(size_t)row * Hdim + col], acc[j][r]);
            }
    }
}

extern "C" void kernel_launch(void* const* d_in, const int* in_sizes, int n_in,
                              void* d_out, int out_size, void* d_ws, size_t ws_size,
                              hipStream_t stream) {
    const float* x  = (const float*)d_in[0];
    const float* Wg = (const float*)d_in[1];
    const float* bg = (const float*)d_in[2];
    const float* W1 = (const float*)d_in[3];
    const float* W3 = (const float*)d_in[4];
    const float* W2 = (const float*)d_in[5];
    float* out = (float*)d_out;

    const size_t cwB   = (size_t)Btok * NE * sizeof(float);         // 128 KB
    const size_t xB    = (size_t)Btok * Hdim * sizeof(unsigned short);  // 8 MB
    const size_t wElem = (size_t)NE * Hdim * Fdim;                  // 33.5M per matrix
    const size_t need  = cwB + xB + 3 * wElem * sizeof(unsigned short); // ~200 MB

    char* ws = (char*)d_ws;
    float* cw = (float*)ws;

    hipMemsetAsync(d_out, 0, (size_t)out_size * sizeof(float), stream);
    moe_gate<<<dim3(Btok / 256), dim3(256), 0, stream>>>(x, Wg, bg, cw);

    dim3 grid(Btok / 64, Fdim / 512, NE);
    if (ws_size >= need) {
        unsigned short* xbf = (unsigned short*)(ws + cwB);
        unsigned short* W1t = (unsigned short*)(ws + cwB + xB);
        unsigned short* W3t = W1t + wElem;
        unsigned short* W2t = W3t + wElem;
        // one-shot bf16 conversion + transpose to K-contiguous layouts
        convert_x<<<dim3((Btok * Hdim) / (256 * 8)), dim3(256), 0, stream>>>(x, xbf);
        transpose_cvt<<<dim3(Fdim / 64, Hdim / 64, NE), dim3(256), 0, stream>>>(W1, W1t, Hdim, Fdim);
        transpose_cvt<<<dim3(Fdim / 64, Hdim / 64, NE), dim3(256), 0, stream>>>(W3, W3t, Hdim, Fdim);
        transpose_cvt<<<dim3(Hdim / 64, Fdim / 64, NE), dim3(256), 0, stream>>>(W2, W2t, Fdim, Hdim);
        moe_ffn_bf16<<<grid, dim3(256), 0, stream>>>(xbf, W1t, W3t, W2t, cw, out);
    } else {
        moe_ffn<<<grid, dim3(256), 0, stream>>>(x, W1, W3, W2, cw, out);
    }
}
